// PillarFusion_31001073943001
// MI455X (gfx1250) — matrix operand order repaired and checked by reading, not compiled, not run on hardware
//
#include <hip/hip_runtime.h>

typedef __attribute__((ext_vector_type(16))) __bf16 v16bf;
typedef __attribute__((ext_vector_type(8)))  __bf16 v8bf;
typedef __attribute__((ext_vector_type(8)))  float  v8f;
typedef __attribute__((ext_vector_type(2)))  float  v2f;

#define N_PTS     100000
#define C_IMG     512
#define C_PT      256
#define M_TILE    64
#define NTHREADS  512            // 16 wave32: 4 M-tiles x 4 N-quarters
#define S_STRIDE  (C_IMG + 8)    // bf16 elems; row = 1040 B (16B multiple, bank-spread)
#define K_CHUNK   32
#define NK        (C_IMG / K_CHUNK)

// ---- Kernel 1: one-time f32 -> bf16 weight conversion into d_ws ----
__global__ __launch_bounds__(256)
void convert_weights_kernel(const float* __restrict__ align_w,
                            __bf16* __restrict__ wbf)
{
    int i = blockIdx.x * 256 + threadIdx.x;
    if (i < C_PT * C_IMG) wbf[i] = (__bf16)align_w[i];
}

// ---- Kernel 2: fused projection + bilinear gather + bf16 WMMA GEMM ----
__global__ __launch_bounds__(NTHREADS)
void pillar_fuse_kernel(const float* __restrict__ point_feat,
                        const float* __restrict__ pillar_centers,
                        const float* __restrict__ img_feat,
                        const float* __restrict__ P2,
                        const float* __restrict__ R0,
                        const float* __restrict__ Tr,
                        const __bf16* __restrict__ wbf,   // [C_PT][C_IMG] bf16, L2-resident
                        const float* __restrict__ align_b,
                        const int*   __restrict__ batch_idx,
                        const int*   __restrict__ img_h_p,
                        const int*   __restrict__ img_w_p,
                        float* __restrict__ out)
{
    __shared__ __attribute__((aligned(16))) __bf16 S_lds[M_TILE * S_STRIDE];
    __shared__ int   off_lds[M_TILE][2];   // row0 / row1 element offsets (x-pair base)
    __shared__ float wgt_lds[M_TILE][4];   // {wy0*pa, wy0*pb, wy1*pa, wy1*pb}

    const int t  = threadIdx.x;
    const int p0 = blockIdx.x * M_TILE;
    const int Wf = img_w_p[0];
    const int Hf = img_h_p[0];
    const int HW = Hf * Wf;

    // ---- Phase 1: projection + bilinear setup (one thread per point) ----
    if (t < M_TILE) {
        int p = p0 + t;
        float w[4] = {0.f, 0.f, 0.f, 0.f};
        int   o[2] = {0, 0};
        if (p < N_PTS) {
            int b = batch_idx[p];
            float px = pillar_centers[p * 3 + 0];
            float py = pillar_centers[p * 3 + 1];
            float pz = pillar_centers[p * 3 + 2];
            const float* Trb = Tr + b * 16;
            const float* R0b = R0 + b * 16;
            const float* P2b = P2 + b * 12;
            float cam[4], rect[4], im[3];
            #pragma unroll
            for (int i = 0; i < 4; ++i)
                cam[i] = Trb[i*4+0]*px + Trb[i*4+1]*py + Trb[i*4+2]*pz + Trb[i*4+3];
            #pragma unroll
            for (int i = 0; i < 4; ++i)
                rect[i] = R0b[i*4+0]*cam[0] + R0b[i*4+1]*cam[1] + R0b[i*4+2]*cam[2] + R0b[i*4+3]*cam[3];
            #pragma unroll
            for (int i = 0; i < 3; ++i)
                im[i] = P2b[i*4+0]*rect[0] + P2b[i*4+1]*rect[1] + P2b[i*4+2]*rect[2] + P2b[i*4+3]*rect[3];
            float depth = fmaxf(im[2], 1e-5f);
            float u = im[0] / depth;
            float v = im[1] / depth;
            bool valid = (im[2] > 0.f) && (u >= 0.f) && (u < (float)Wf) &&
                         (v >= 0.f) && (v < (float)Hf);
            if (valid) {
                float x0f = floorf(u), y0f = floorf(v);
                float wx1 = u - x0f, wx0 = 1.f - wx1;
                float wy1 = v - y0f, wy0 = 1.f - wy1;
                int x0 = (int)x0f;            // in [0, W-1] given valid
                int y0 = (int)y0f;            // in [0, H-1] given valid
                // x-pair base + shifted weights so the b64 never crosses the row end
                int   xa = (x0 >= Wf - 1) ? (Wf - 2) : x0;
                float pa = (x0 >= Wf - 1) ? 0.f : wx0;
                float pb = (x0 >= Wf - 1) ? wx0 : wx1;
                // row1 clamped; its weight zeroed when y1 is out of bounds
                int   yb   = (y0 + 1 > Hf - 1) ? (Hf - 1) : (y0 + 1);
                float wy1e = (y0 + 1 > Hf - 1) ? 0.f : wy1;
                int base = b * C_IMG * HW;
                o[0] = base + y0 * Wf + xa;
                o[1] = base + yb * Wf + xa;
                w[0] = wy0 * pa; w[1] = wy0 * pb;
                w[2] = wy1e * pa; w[3] = wy1e * pb;
            }
        }
        off_lds[t][0] = o[0]; off_lds[t][1] = o[1];
        #pragma unroll
        for (int ci = 0; ci < 4; ++ci) wgt_lds[t][ci] = w[ci];
    }
    __syncthreads();

    // ---- Phase 2: bilinear gather, 2 x b64 per sample -> S_lds (bf16) ----
    for (int idx = t; idx < M_TILE * C_IMG; idx += NTHREADS) {
        int p = idx / C_IMG;
        int c = idx % C_IMG;
        int cb = c * HW;
        v2f f0 = *(const v2f*)(img_feat + off_lds[p][0] + cb);  // (F[y0][xa], F[y0][xa+1])
        v2f f1 = *(const v2f*)(img_feat + off_lds[p][1] + cb);  // (F[y1][xa], F[y1][xa+1])
        float s = wgt_lds[p][0] * f0.x + wgt_lds[p][1] * f0.y
                + wgt_lds[p][2] * f1.x + wgt_lds[p][3] * f1.y;
        S_lds[p * S_STRIDE + c] = (__bf16)s;
    }
    __syncthreads();   // the only barrier before the tensor-core phase

    // ---- Phase 3: GEMM via v_wmma_f32_16x16x32_bf16 ----
    const int wave  = t >> 5;
    const int lane  = t & 31;
    const int mt    = wave & 3;   // which 16-point tile (0..3)
    const int nq    = wave >> 2;  // which 64-out-channel quarter (0..3)
    const int lhalf = lane >> 4;  // 0 | 1
    const int l15   = lane & 15;

    v8f acc[4];
    // C init = point_feat + align_b (epilogue folded into accumulator)
    #pragma unroll
    for (int nt = 0; nt < 4; ++nt) {
        #pragma unroll
        for (int r = 0; r < 8; ++r) {
            int prow = p0 + mt * 16 + r + lhalf * 8;   // C/D layout: VGPR r, lane half -> M
            int col  = nq * 64 + nt * 16 + l15;        // lane%16 -> N
            float vv = 0.f;
            if (prow < N_PTS) vv = point_feat[prow * C_PT + col] + align_b[col];
            acc[nt][r] = vv;
        }
    }

    // Per-wave B-fragment base rows (constant across K)
    const __bf16* wrow0 = wbf + (nq * 64 + 0 * 16 + l15) * C_IMG;
    const __bf16* wrow1 = wbf + (nq * 64 + 1 * 16 + l15) * C_IMG;
    const __bf16* wrow2 = wbf + (nq * 64 + 2 * 16 + l15) * C_IMG;
    const __bf16* wrow3 = wbf + (nq * 64 + 3 * 16 + l15) * C_IMG;
    const __bf16* wrows[4] = {wrow0, wrow1, wrow2, wrow3};

    for (int k = 0; k < NK; ++k) {
        // A fragment (16x32 bf16 ISA layout): lane half selects K runs {0..7,16..23}/{8..15,24..31}
        const __bf16* ap = &S_lds[(mt * 16 + l15) * S_STRIDE + k * K_CHUNK + lhalf * 8];
        v8bf alo = *(const v8bf*)(ap);        // K = base + lhalf*8 + 0..7
        v8bf ahi = *(const v8bf*)(ap + 16);   // K = base + 16 + lhalf*8 + 0..7
        v16bf afrag = __builtin_shufflevector(alo, ahi,
            0, 1, 2, 3, 4, 5, 6, 7, 8, 9, 10, 11, 12, 13, 14, 15);

        const int kcol = k * K_CHUNK + lhalf * 8;   // 16B-aligned bf16 column offset
        #pragma unroll
        for (int nt = 0; nt < 4; ++nt) {
            // B fragment (32x16 bf16): lane -> N = lane%16, K runs {0..7,16..23}/{8..15,24..31} by lane half
            const __bf16* bp = wrows[nt] + kcol;
            v8bf blo = *(const v8bf*)(bp);        // global_load_b128 (L2 hit)
            v8bf bhi = *(const v8bf*)(bp + 16);
            v16bf bfrag = __builtin_shufflevector(blo, bhi,
                0, 1, 2, 3, 4, 5, 6, 7, 8, 9, 10, 11, 12, 13, 14, 15);
            acc[nt] = __builtin_amdgcn_wmma_f32_16x16x32_bf16(
                false, afrag, false, bfrag, (short)0, acc[nt], false, false);
        }
    }

    // ---- Phase 4: store (out = point_feat + align_b + sampled @ W^T, already in acc) ----
    #pragma unroll
    for (int nt = 0; nt < 4; ++nt) {
        #pragma unroll
        for (int r = 0; r < 8; ++r) {
            int prow = p0 + mt * 16 + r + lhalf * 8;
            int col  = nq * 64 + nt * 16 + l15;
            if (prow < N_PTS) out[prow * C_PT + col] = acc[nt][r];
        }
    }
}

extern "C" void kernel_launch(void* const* d_in, const int* in_sizes, int n_in,
                              void* d_out, int out_size, void* d_ws, size_t ws_size,
                              hipStream_t stream) {
    const float* point_feat     = (const float*)d_in[0];
    const float* pillar_centers = (const float*)d_in[1];
    const float* img_feat       = (const float*)d_in[2];
    const float* P2             = (const float*)d_in[3];
    const float* R0             = (const float*)d_in[4];
    const float* Tr             = (const float*)d_in[5];
    const float* align_w        = (const float*)d_in[6];
    const float* align_b        = (const float*)d_in[7];
    const int*   batch_idx      = (const int*)d_in[8];
    const int*   img_h          = (const int*)d_in[9];
    const int*   img_w          = (const int*)d_in[10];
    float* out = (float*)d_out;

    __bf16* wbf = (__bf16*)d_ws;   // 256 KB bf16 weight copy in scratch

    int cblocks = (C_PT * C_IMG + 255) / 256;  // 512
    hipLaunchKernelGGL(convert_weights_kernel, dim3(cblocks), dim3(256), 0, stream,
                       align_w, wbf);

    int nblocks = (N_PTS + M_TILE - 1) / M_TILE;  // 1563
    hipLaunchKernelGGL(pillar_fuse_kernel, dim3(nblocks), dim3(NTHREADS), 0, stream,
                       point_feat, pillar_centers, img_feat, P2, R0, Tr,
                       wbf, align_b, batch_idx, img_h, img_w, out);
}
